// ViscoplasticMaterialModel_28578712388158
// MI455X (gfx1250) — matrix operand order repaired and checked by reading, not compiled, not run on hardware
//
#include <hip/hip_runtime.h>
#include <hip/hip_bf16.h>
#include <math.h>

typedef __attribute__((ext_vector_type(16))) _Float16 v16h;
typedef __attribute__((ext_vector_type(8)))  float    v8f;

#define MAX_ROUNDS 7936   // >= 256 timesteps * 30 inner iterations

#if __has_builtin(__builtin_amdgcn_tensor_load_to_lds)
#define HAVE_TDM 1
#else
#define HAVE_TDM 0
#endif

__device__ __forceinline__ float gelu_f(float x) {
  float x3 = x * x * x;
  return 0.5f * x * (1.0f + tanhf(0.7978845608028654f * (x + 0.044715f * x3)));
}
__device__ __forceinline__ float softplus_f(float x) {
  return (x > 20.f) ? x : log1pf(__expf(x));
}
__device__ __forceinline__ float sigmoid_f(float x) {
  return 1.0f / (1.0f + __expf(-x));
}

__device__ __forceinline__ v8f wmma_f16(v16h a, v16h b, v8f c) {
  // 8 args: (neg_a, A, neg_b, B, c_mod, C, reuse_a, reuse_b)
  return __builtin_amdgcn_wmma_f32_16x16x32_f16(false, a, false, b, (short)0, c, false, false);
}

// ---------------------------------------------------------------------------
// Tensor Data Mover: 1-D tile of f32 from global -> LDS (ISA 08, D# §8.3/8.4).
// group0: count=1 | lds_addr | global_addr[56:0] | type=2
// group1: data_size=2(4B); tensor_dim0=tile_dim0=nelem; tensor_dim1=1;
//         tensor_dim0_stride=nelem. Groups 2/3 zero (<=2D tensor).
// lds_addr: generic LDS pointer low 32 bits == workgroup LDS byte address.
// ---------------------------------------------------------------------------
__device__ __forceinline__ void tdm_load_f32(const float* gsrc, void* ldst, int nelem) {
#if HAVE_TDM
  typedef __attribute__((ext_vector_type(4))) unsigned u32x4;
  typedef __attribute__((ext_vector_type(8))) int i32x8;
  typedef __attribute__((ext_vector_type(4))) int i32x4;
  unsigned long long ga = (unsigned long long)gsrc;
  unsigned ldsa = (unsigned)(unsigned long long)ldst;
  u32x4 g0 = {0u, 0u, 0u, 0u};
  g0[0] = 1u;                                                  // count=1 (valid user D#)
  g0[1] = ldsa;                                                // lds_addr
  g0[2] = (unsigned)ga;                                        // global_addr[31:0]
  g0[3] = (unsigned)((ga >> 32) & 0x01FFFFFFull) | (2u << 30); // addr[56:32] | type=2
  i32x8 g1 = {0, 0, 0, 0, 0, 0, 0, 0};
  g1[0] = (2 << 16);                       // data_size = 4 bytes
  g1[1] = (nelem & 0xFFFF) << 16;          // tensor_dim0 lo16 (bits 63:48)
  g1[2] = (nelem >> 16) | (1 << 16);       // tensor_dim0 hi16 | tensor_dim1=1 lo16
  g1[3] = (nelem & 0xFFFF) << 16;          // tile_dim0 (bits 127:112)
  g1[4] = 0;                               // tile_dim1=0, tile_dim2=0 (1-D tile)
  g1[5] = nelem;                           // tensor_dim0_stride lo32
  i32x4 z4 = {0, 0, 0, 0};
#if defined(__clang_major__) && __clang_major__ >= 23
  i32x8 z8 = {0, 0, 0, 0, 0, 0, 0, 0};
  __builtin_amdgcn_tensor_load_to_lds(g0, g1, z4, z4, z8, 0);
#else
  __builtin_amdgcn_tensor_load_to_lds(g0, g1, z4, z4, 0);
#endif
#else
  (void)gsrc; (void)ldst; (void)nelem;
#endif
}

__device__ __forceinline__ void tdm_wait() {
#if __has_builtin(__builtin_amdgcn_s_wait_tensorcnt)
  __builtin_amdgcn_s_wait_tensorcnt(0);
#endif
}

// A fragment (16x32 f16) from row-major f16 LDS: base[row*stride + k]
// wave32 layout: lanes 0-15 row r, V0-3 K 0..7, V4-7 K 16..23; lanes 16-31: +8
__device__ __forceinline__ v16h load_fragA(const _Float16* base, int stride, int row0, int k0, int lane) {
  union { v16h v; unsigned u[8]; } r;
  const int row = row0 + (lane & 15);
  const _Float16* p = base + row * stride + k0 + 8 * (lane >> 4);
  #pragma unroll
  for (int i = 0; i < 8; ++i) {
    int k = (i < 4) ? (2 * i) : (16 + 2 * (i - 4));
    r.u[i] = *(const unsigned*)(p + k);
  }
  return r.v;
}

// B fragment (32x16 f16) from TRANSPOSED row-major f16 LDS BT[n][k]
// lanes 0-15 column n: V0-3 K 0..7, V4-7 K 16..23; lanes 16-31: +8
__device__ __forceinline__ v16h load_fragB(const _Float16* baseT, int stride, int col0, int k0, int lane) {
  union { v16h v; unsigned u[8]; } r;
  const int col = col0 + (lane & 15);
  const _Float16* p = baseT + col * stride + k0 + 8 * (lane >> 4);
  #pragma unroll
  for (int i = 0; i < 8; ++i) r.u[i] = *(const unsigned*)(p + ((i < 4) ? (2 * i) : (16 + 2 * (i - 4))));
  return r.v;
}

// ---------------------------------------------------------------------------
// Fused 1D FNO: one workgroup per batch element. All 3 layers + head in LDS.
// ---------------------------------------------------------------------------
struct FnoP {
  const float* in0; const float* in1;
  const float* lift; const float* fc1; const float* bfc1; const float* fc2; const float* bfc2;
  const float* pw0; const float* pw1; const float* pw2;
  const float* sr0; const float* sr1; const float* sr2;
  const float* si0; const float* si1; const float* si2;
  float* mf;
  int din; int modes;
};

__global__ __launch_bounds__(256) void fno_kernel(FnoP P) {
  __shared__ _Float16 h16c[32][128];   // h, channel-major (DFT A)
  __shared__ _Float16 h16t[128][32];   // h, l-major (pointwise A)
  __shared__ float    accs[32][128];   // f32 accumulator (spec + pw)
  __shared__ _Float16 WfT[16][128];    // forward-DFT B, transposed [m'][l]
  __shared__ _Float16 WiT[128][32];    // inverse-DFT B, transposed [l][k]
  __shared__ _Float16 pwT[32][32];     // pointwise weights transposed [o][i]
  __shared__ float    pwStage[2][1024];// TDM double-buffer for pw weights
  __shared__ float xfr[32][8];
  __shared__ float xfi[32][8];
  __shared__ _Float16 om16[32][32];    // iDFT A (K padded to 32)
  __shared__ float feat[32];
  __shared__ float g1[64];

  const int tid  = threadIdx.x;
  const int lane = tid & 31;
  const int wave = tid >> 5;
  const int b    = blockIdx.x;

  const float* pws[3] = {P.pw0, P.pw1, P.pw2};
  const float* srs[3] = {P.sr0, P.sr1, P.sr2};
  const float* sis[3] = {P.si0, P.si1, P.si2};

  // Kick off async DMA of layer-0 pointwise weights; overlaps twiddles + lift.
  if (HAVE_TDM && wave == 0) {
    tdm_load_f32(pws[0], &pwStage[0][0], 1024);
  }

  // twiddles (mode-independent; unused modes are zeroed in om16)
  for (int idx = tid; idx < 16 * 128; idx += 256) {
    int m = idx >> 7, l = idx & 127;
    float th = 6.2831853071795864f * (float)((m & 7) * l) / 128.0f;
    WfT[m][l] = (_Float16)((m < 8) ? __cosf(th) : -__sinf(th));
  }
  for (int idx = tid; idx < 128 * 32; idx += 256) {
    int l = idx >> 5, k = idx & 31;
    float v = 0.f;
    if (k < 8) {
      float c = (k == 0 ? 1.f : 2.f) / 128.f;
      v = c * __cosf(6.2831853071795864f * (float)(k * l) / 128.f);
    } else if (k < 16) {
      int m = k - 8;
      float c = (m == 0 ? 1.f : 2.f) / 128.f;
      v = -c * __sinf(6.2831853071795864f * (float)(m * l) / 128.f);
    }
    WiT[l][k] = (_Float16)v;
  }

  // lift: h[c][l] = sum_i x[i][l] * lift[i][c]
  for (int idx = tid; idx < 32 * 128; idx += 256) {
    int c = idx >> 7, l = idx & 127;
    float x0 = (P.din == 1) ? P.in0[b * 128 + l] : (1.0f / P.in0[b * 128 + l]);
    float v = x0 * P.lift[c];
    if (P.din == 2) v += P.in1[b * 128 + l] * P.lift[32 + c];
    h16c[c][l] = (_Float16)v;
    h16t[l][c] = (_Float16)v;
  }
  __syncthreads();

  for (int layer = 0; layer < 3; ++layer) {
    const int slot = layer & 1;
    // 0) weights: wait for this layer's DMA, enqueue next layer's, convert f32->f16^T
    if (HAVE_TDM) {
      if (wave == 0) {
        tdm_wait();
        if (layer < 2) tdm_load_f32(pws[layer + 1], &pwStage[slot ^ 1][0], 1024);
      }
    } else {
      const float* pw = pws[layer];
      for (int idx = tid; idx < 1024; idx += 256) pwStage[slot][idx] = pw[idx];
    }
    __syncthreads();
    for (int idx = tid; idx < 1024; idx += 256) {
      int i = idx >> 5, o = idx & 31;
      pwT[o][i] = (_Float16)pwStage[slot][idx];
    }
    __syncthreads();

    // 1) pointwise GEMM: acc[o][l] = sum_i h[i][l]*pw[i][o]  (M=l:128, N=o:32, K=32)
    #pragma unroll
    for (int r = 0; r < 2; ++r) {
      int job = wave + 8 * r;
      int mt = job >> 1, nt = job & 1;
      v8f c = {};
      v16h a   = load_fragA(&h16t[0][0], 32, mt * 16, 0, lane);
      v16h bf  = load_fragB(&pwT[0][0], 32, nt * 16, 0, lane);
      c = wmma_f16(a, bf, c);
      int col = nt * 16 + (lane & 15);
      int rbase = mt * 16 + 8 * (lane >> 4);
      #pragma unroll
      for (int i = 0; i < 8; ++i) accs[col][rbase + i] = c[i];  // transpose into [o][l]
    }
    __syncthreads();

    // 2) forward DFT, modes 0..7: xf[ch][m']  (M=ch:32, N=16, K=128)
    if (wave < 2) {
      int mt = wave;
      v8f c = {};
      #pragma unroll
      for (int kk = 0; kk < 4; ++kk) {
        v16h a  = load_fragA(&h16c[0][0], 128, mt * 16, kk * 32, lane);
        v16h bf = load_fragB(&WfT[0][0], 128, 0, kk * 32, lane);
        c = wmma_f16(a, bf, c);
      }
      int col = lane & 15;
      int rbase = mt * 16 + 8 * (lane >> 4);
      #pragma unroll
      for (int i = 0; i < 8; ++i) {
        if (col < 8) xfr[rbase + i][col] = c[i];
        else         xfi[rbase + i][col - 8] = c[i];
      }
    }
    __syncthreads();

    // 3) spectral multiply: om[o][m] = sum_i xf[i][m] * (sr + i*si)[i][o][m]
    {
      int o = tid & 31, m = tid >> 5;  // m in 0..7
      float orr = 0.f, oii = 0.f;
      if (m < P.modes) {
        const float* sr = srs[layer];
        const float* si = sis[layer];
        #pragma unroll 4
        for (int i = 0; i < 32; ++i) {
          float xr = xfr[i][m], xm = xfi[i][m];
          float wr = sr[(i * 32 + o) * 8 + m];
          float wi = si[(i * 32 + o) * 8 + m];
          orr += xr * wr - xm * wi;
          oii += xr * wi + xm * wr;
        }
      }
      om16[o][m]      = (_Float16)orr;
      om16[o][8 + m]  = (_Float16)oii;
      om16[o][16 + m] = (_Float16)0.f;
      om16[o][24 + m] = (_Float16)0.f;
    }
    __syncthreads();

    // 4) inverse DFT: acc[ch][l] += om(32x32pad) * Wi(32x128)
    #pragma unroll
    for (int r = 0; r < 2; ++r) {
      int job = wave + 8 * r;
      int mt = job & 1, nt = job >> 1;
      v8f c = {};
      v16h a  = load_fragA(&om16[0][0], 32, mt * 16, 0, lane);
      v16h bf = load_fragB(&WiT[0][0], 32, nt * 16, 0, lane);
      c = wmma_f16(a, bf, c);
      int col = nt * 16 + (lane & 15);
      int rbase = mt * 16 + 8 * (lane >> 4);
      #pragma unroll
      for (int i = 0; i < 8; ++i) accs[rbase + i][col] += c[i];
    }
    __syncthreads();

    // 5) GELU -> next-layer h (both layouts)
    for (int idx = tid; idx < 32 * 128; idx += 256) {
      int c = idx >> 7, l = idx & 127;
      float v = gelu_f(accs[c][l]);
      h16c[c][l] = (_Float16)v;
      h16t[l][c] = (_Float16)v;
    }
    __syncthreads();
  }

  // head: mean over l -> fc1 -> gelu -> fc2
  if (tid < 32) {
    float s = 0.f;
    for (int l = 0; l < 128; ++l) s += (float)h16c[tid][l];
    feat[tid] = s * (1.0f / 128.0f);
  }
  __syncthreads();
  if (tid < 64) {
    float s = P.bfc1[tid];
    #pragma unroll 8
    for (int c = 0; c < 32; ++c) s += feat[c] * P.fc1[c * 64 + tid];
    g1[tid] = gelu_f(s);
  }
  __syncthreads();
  if (tid == 0) {
    float s = P.bfc2[0];
    for (int o = 0; o < 64; ++o) s += g1[o] * P.fc2[o];
    P.mf[b] = s;
  }
}

// ---------------------------------------------------------------------------
// Per-batch solver constants
// ---------------------------------------------------------------------------
__global__ __launch_bounds__(64) void precomp_kernel(
    const float* W1, const float* W2, const float* b1,
    const float* Wctx, const float* Wqv, const float* Wv, const float* Wz, const float* wout,
    const float* mf1, const float* mf2,
    float* c1, float* sbv, float* az, float* consts) {
  __shared__ float v[50];
  __shared__ float red[64];
  int b = blockIdx.x, t = threadIdx.x;
  float m2 = mf2[b];
  if (t < 50) v[t] = softplus_f(m2 * Wctx[t]);
  __syncthreads();
  red[t] = (t < 50) ? v[t] * Wqv[t] : 0.f;
  __syncthreads();
  for (int s = 32; s > 0; s >>= 1) { if (t < s) red[t] += red[t + s]; __syncthreads(); }
  if (t == 0) sbv[b] = softplus_f(red[0]);
  float m1 = mf1[b];
  for (int z = t; z < 50; z += 64) {
    float a = 0.f;
    for (int k = 0; k < 50; ++k) a += v[k] * Wv[k * 50 + z];
    az[b * 50 + z] = a;
    c1[b * 50 + z] = m1 * W1[100 + z] + b1[z];
  }
  if (b == 0) {
    for (int j = t; j < 50; j += 64) {
      consts[j]       = W1[j];               // a0 (d/de row)
      consts[50 + j]  = W1[50 + j];          // a1 (d/dp row)
      consts[100 + j] = W1[50 + j] * W2[j];  // e1 = a1*W2
      consts[150 + j] = W1[j] * W2[j];       // e0 = a0*W2 (stress)
      float sz = softplus_f(Wz[j]);
      consts[200 + j] = sz;                                // swz
      consts[250 + j] = sz * softplus_f(wout[j]);          // zz
    }
  }
}

__global__ void init_ws(unsigned* barCnt, int n) {
  int i = blockIdx.x * blockDim.x + threadIdx.x;
  if (i < n) barCnt[i] = 0u;
}

// ---------------------------------------------------------------------------
// Persistent solver: 16 blocks x 256 threads = 4096 (one batch element each).
// Global convergence test replicated bitwise-identically on every thread via
// fixed-order cross-block summation + monotone-round software grid barrier.
// ---------------------------------------------------------------------------
__global__ __launch_bounds__(256) void solver_kernel(
    const float* e, const float* c1, const float* sbv, const float* az,
    const float* consts, float* blockSums, unsigned* barCnt, float* xi_out) {
  __shared__ float a0s[50], a1s[50], e1s[50], swzs[50], zzs[50];
  __shared__ float redD[256], redX[256];
  const int tid = threadIdx.x;
  const int b = blockIdx.x * 256 + tid;
  for (int j = tid; j < 50; j += 256) {
    a0s[j] = consts[j]; a1s[j] = consts[50 + j]; e1s[j] = consts[100 + j];
    swzs[j] = consts[200 + j]; zzs[j] = consts[250 + j];
  }
  float c1r[50], azr[50];
  #pragma unroll
  for (int j = 0; j < 50; ++j) { c1r[j] = c1[b * 50 + j]; azr[j] = az[b * 50 + j]; }
  const float sbb = sbv[b];
  __syncthreads();

  const float* eb = e + b * 256;
  float xi = 0.f;
  xi_out[b * 257] = 0.f;
  int round = 0;

  for (int t = 0; t < 256; ++t) {
    float ei = eb[t];
    __builtin_prefetch(eb + t + 1, 0, 1);
    float d = (t == 0) ? eb[1] : ((t < 255) ? (eb[t + 1] - eb[t]) : (eb[255] - eb[254]));
    float xc = xi + 0.5f * d;
    float err = 1.0f;
    int cnt = 1;
    for (;;) {
      // g = d/dp [ energy(e_i, p, mf1) ] + d/dp [ DT * diss((p-q)/DT, mf2) ]
      float ge = 0.f;
      #pragma unroll
      for (int j = 0; j < 50; ++j) {
        float pre = ei * a0s[j] + xc * a1s[j] + c1r[j];
        ge += sigmoid_f(pre) * e1s[j];
      }
      float qt = (xc - xi) * 100.0f * sbb;  // q_tilde * s_b  (1/DT = 100)
      float gd = 0.f;
      #pragma unroll
      for (int z = 0; z < 50; ++z) gd += sigmoid_f(qt * swzs[z] + azr[z]) * zzs[z];
      gd *= sbb;
      float g = ge + gd;
      float xn = xc - 0.001f * g / (float)cnt;
      float dl = xn - xc;
      redD[tid] = dl * dl; redX[tid] = xc * xc;
      __syncthreads();
      #pragma unroll
      for (int s = 128; s > 0; s >>= 1) {
        if (tid < s) { redD[tid] += redD[tid + s]; redX[tid] += redX[tid + s]; }
        __syncthreads();
      }
      if (tid == 0) {
        __hip_atomic_store(&blockSums[round * 32 + blockIdx.x * 2 + 0], redD[0],
                           __ATOMIC_RELEASE, __HIP_MEMORY_SCOPE_AGENT);
        __hip_atomic_store(&blockSums[round * 32 + blockIdx.x * 2 + 1], redX[0],
                           __ATOMIC_RELEASE, __HIP_MEMORY_SCOPE_AGENT);
        __hip_atomic_fetch_add(&barCnt[round], 1u, __ATOMIC_ACQ_REL, __HIP_MEMORY_SCOPE_AGENT);
        while (__hip_atomic_load(&barCnt[round], __ATOMIC_ACQUIRE, __HIP_MEMORY_SCOPE_AGENT) < 16u)
          __builtin_amdgcn_s_sleep(2);
      }
      __syncthreads();
      float D2 = 0.f, X2 = 0.f;
      #pragma unroll
      for (int k = 0; k < 16; ++k) {  // fixed order -> deterministic & uniform
        D2 += __hip_atomic_load(&blockSums[round * 32 + k * 2 + 0], __ATOMIC_ACQUIRE, __HIP_MEMORY_SCOPE_AGENT);
        X2 += __hip_atomic_load(&blockSums[round * 32 + k * 2 + 1], __ATOMIC_ACQUIRE, __HIP_MEMORY_SCOPE_AGENT);
      }
      ++round;
      float errn = (cnt > 1) ? (sqrtf(D2) / (sqrtf(X2) + 1e-8f)) : err;
      xc = xn;
      ++cnt;
      if (!(errn > 0.001f && cnt <= 30)) break;
      err = errn;
    }
    xi = xc;
    xi_out[b * 257 + t + 1] = xi;
  }
}

// ---------------------------------------------------------------------------
// stress[b,t] = sum_j sigmoid(e*a0 + xi*a1 + c1) * a0[j]*W2[j]
// ---------------------------------------------------------------------------
__global__ __launch_bounds__(256) void stress_kernel(
    const float* e, const float* c1, const float* consts, const float* xi, float* stress) {
  __shared__ float a0s[50], a1s[50], e0s[50], c1s[50];
  int b = blockIdx.x, t = threadIdx.x;
  for (int j = t; j < 50; j += 256) {
    a0s[j] = consts[j]; a1s[j] = consts[50 + j]; e0s[j] = consts[150 + j];
    c1s[j] = c1[b * 50 + j];
  }
  __syncthreads();
  float ei = e[b * 256 + t];
  float p = xi[b * 257 + t];  // xi[:, :-1]
  float s = 0.f;
  #pragma unroll
  for (int j = 0; j < 50; ++j) {
    float pre = ei * a0s[j] + p * a1s[j] + c1s[j];
    s += sigmoid_f(pre) * e0s[j];
  }
  stress[b * 256 + t] = s;
}

extern "C" void kernel_launch(void* const* d_in, const int* in_sizes, int n_in,
                              void* d_out, int out_size, void* d_ws, size_t ws_size,
                              hipStream_t stream) {
  (void)in_sizes; (void)n_in; (void)out_size; (void)ws_size;
  // Top-level dict order; nested param dicts flattened in jax-pytree (sorted-key) order.
  const float* e     = (const float*)d_in[0];
  const float* E     = (const float*)d_in[1];
  const float* Y     = (const float*)d_in[2];
  const float* edot0 = (const float*)d_in[4];   // d_in[3] = n (unused by reference)
  const float* W1 = (const float*)d_in[5];
  const float* W2 = (const float*)d_in[6];
  const float* b1 = (const float*)d_in[7];      // d_in[8] = b2 (drops out of grads)
  const float* Wctx = (const float*)d_in[9];
  const float* Wqv  = (const float*)d_in[10];
  const float* Wv   = (const float*)d_in[11];
  const float* Wz   = (const float*)d_in[12];
  const float* wout = (const float*)d_in[13];

  float* ws = (float*)d_ws;                 // total ~2.8 MB of scratch used
  float* mf1 = ws;                          // 4096
  float* mf2 = mf1 + 4096;                  // 4096
  float* c1  = mf2 + 4096;                  // 4096*50
  float* sbv = c1 + 4096 * 50;              // 4096
  float* az  = sbv + 4096;                  // 4096*50
  float* consts = az + 4096 * 50;           // 512 (300 used)
  float* blockSums = consts + 512;          // MAX_ROUNDS*32
  unsigned* barCnt = (unsigned*)(blockSums + MAX_ROUNDS * 32);  // MAX_ROUNDS

  init_ws<<<(MAX_ROUNDS + 255) / 256, 256, 0, stream>>>(barCnt, MAX_ROUNDS);

  FnoP p1{};
  p1.in0 = E; p1.in1 = nullptr;
  p1.bfc1 = (const float*)d_in[14]; p1.bfc2 = (const float*)d_in[15];
  p1.fc1  = (const float*)d_in[16]; p1.fc2  = (const float*)d_in[17];
  p1.lift = (const float*)d_in[18];
  p1.pw0 = (const float*)d_in[19]; p1.pw1 = (const float*)d_in[20]; p1.pw2 = (const float*)d_in[21];
  p1.si0 = (const float*)d_in[22]; p1.si1 = (const float*)d_in[23]; p1.si2 = (const float*)d_in[24];
  p1.sr0 = (const float*)d_in[25]; p1.sr1 = (const float*)d_in[26]; p1.sr2 = (const float*)d_in[27];
  p1.mf = mf1; p1.din = 1; p1.modes = 8;
  fno_kernel<<<4096, 256, 0, stream>>>(p1);

  FnoP p2{};
  p2.in0 = Y; p2.in1 = edot0;
  p2.bfc1 = (const float*)d_in[28]; p2.bfc2 = (const float*)d_in[29];
  p2.fc1  = (const float*)d_in[30]; p2.fc2  = (const float*)d_in[31];
  p2.lift = (const float*)d_in[32];
  p2.pw0 = (const float*)d_in[33]; p2.pw1 = (const float*)d_in[34]; p2.pw2 = (const float*)d_in[35];
  p2.si0 = (const float*)d_in[36]; p2.si1 = (const float*)d_in[37]; p2.si2 = (const float*)d_in[38];
  p2.sr0 = (const float*)d_in[39]; p2.sr1 = (const float*)d_in[40]; p2.sr2 = (const float*)d_in[41];
  p2.mf = mf2; p2.din = 2; p2.modes = 4;
  fno_kernel<<<4096, 256, 0, stream>>>(p2);

  precomp_kernel<<<4096, 64, 0, stream>>>(W1, W2, b1, Wctx, Wqv, Wv, Wz, wout,
                                          mf1, mf2, c1, sbv, az, consts);

  float* out_stress = (float*)d_out;
  float* out_xi = out_stress + 4096 * 256;
  solver_kernel<<<16, 256, 0, stream>>>(e, c1, sbv, az, consts, blockSums, barCnt, out_xi);
  stress_kernel<<<4096, 256, 0, stream>>>(e, c1, consts, out_xi, out_stress);
}
